// Quantizer_74431783239915
// MI455X (gfx1250) — matrix operand order repaired and checked by reading, not compiled, not run on hardware
//
#include <hip/hip_runtime.h>
#include <hip/hip_bf16.h>

// ---------------------------------------------------------------------------
// VQ quantizer for MI455X (gfx1250, wave32, WMMA).
//   distance(n,k) = x_sq[n] - 2 * (X @ W^T)[n,k] + w_sq[k];  out = W[argmin]
// Cross term via V_WMMA_F32_16X16X32_BF16 (f32 accumulate).
//   K_codes = 256, D = 1000 (padded to 1024), N = 65536.
// Workspace layout: [0, 512KB) Wb bf16[256][1024]; [512KB, +1KB) w_sq f32[256].
//
// CDNA5-specific paths used:
//   - v_wmma_f32_16x16x32_bf16 (16 N-tiles x 8 M-rows of acc per wave)
//   - global_load_async_to_lds_b128 (ASYNCcnt) double-buffered B staging
//   - s_wait_asynccnt pipelined against A-tile f32->bf16 conversion
//   - sched_group_barrier pipeline hints to overlap ds_load with WMMA
//   - wave32 width-16 shuffle argmin matching the 16x16 C/D lane layout
// ---------------------------------------------------------------------------

#define KCODES 256
#define DIM    1000
#define DPAD   1024
#define TM     128   // rows of X per block
#define KC     64    // k-chunk staged in LDS per iteration

typedef __bf16  bf16_t;
typedef bf16_t  v16bf __attribute__((ext_vector_type(16)));
typedef float   v8f   __attribute__((ext_vector_type(8)));
typedef float   v4f   __attribute__((ext_vector_type(4)));   // clang-native float4

#if __has_builtin(__builtin_amdgcn_global_load_async_to_lds_b128) && \
    __has_builtin(__builtin_amdgcn_s_wait_asynccnt)
#define USE_ASYNC_LDS 1
// Probe-derived signature: (v4i __device__*, v4i __shared__*, imm offset, imm cpol)
typedef int v4i_vs __attribute__((__vector_size__(16)));
typedef __attribute__((address_space(1))) v4i_vs glob_v4i;
typedef __attribute__((address_space(3))) v4i_vs lds_v4i;
#else
#define USE_ASYNC_LDS 0
#endif

#if __has_builtin(__builtin_amdgcn_sched_group_barrier)
#define SGB(mask, size, id) __builtin_amdgcn_sched_group_barrier((mask), (size), (id))
#else
#define SGB(mask, size, id)
#endif
// sched_group_barrier masks: 0x008 = MFMA/WMMA, 0x100 = DS read.

union FragU {
    uint4 u4[2];
    v16bf v;
};

__device__ __forceinline__ unsigned short f2bf(float f) {
    unsigned int u = __builtin_bit_cast(unsigned int, f);
    u += 0x7FFFu + ((u >> 16) & 1u);        // round-to-nearest-even
    return (unsigned short)(u >> 16);
}

template <int N>
__device__ __forceinline__ void wait_async() {
#if USE_ASYNC_LDS
    __builtin_amdgcn_s_wait_asynccnt(N);
#endif
}

// Stage one 256x64 bf16 W-tile (32 KB): 8 x async b128 per thread.
__device__ __forceinline__ void stage_B(const unsigned short* __restrict__ Wb,
                                        unsigned short* dst, int tid, int kc) {
    const unsigned short* gsrc = Wb + (size_t)tid * DPAD + kc;
    unsigned short* ldst = dst + tid * KC;
#if USE_ASYNC_LDS
    #pragma unroll
    for (int i = 0; i < 8; ++i)
        __builtin_amdgcn_global_load_async_to_lds_b128(
            (glob_v4i*)(gsrc + i * 8), (lds_v4i*)(ldst + i * 8), 0, 0);
#else
    #pragma unroll
    for (int i = 0; i < 8; ++i)
        ((uint4*)ldst)[i] = ((const uint4*)gsrc)[i];
#endif
}

// --- pass 1: W (f32) -> Wb (bf16, padded to DPAD) and w_sq --------------------
__global__ void prep_w(const float* __restrict__ W,
                       unsigned short* __restrict__ Wb,
                       float* __restrict__ wsq) {
    const int k = blockIdx.x;     // code row 0..255
    const int t = threadIdx.x;    // 0..255
    __shared__ float red[256];
    float s = 0.f;
    for (int d = t; d < DPAD; d += 256) {
        float v = (d < DIM) ? W[(size_t)k * DIM + d] : 0.f;
        Wb[(size_t)k * DPAD + d] = f2bf(v);
        s += v * v;
    }
    red[t] = s;
    __syncthreads();
    for (int off = 128; off > 0; off >>= 1) {
        if (t < off) red[t] += red[t + off];
        __syncthreads();
    }
    if (t == 0) wsq[k] = red[0];
}

// --- pass 2: bf16 WMMA GEMM + distance argmin + gather -----------------------
__global__ __launch_bounds__(256, 1)
void vq_kernel(const float* __restrict__ X,
               const float* __restrict__ W,
               const unsigned short* __restrict__ Wb,
               const float* __restrict__ wsq,
               float* __restrict__ out) {
    __shared__ unsigned short Alds[TM * KC];           // 16 KB: X tile (bf16)
    __shared__ unsigned short Blds[2][KCODES * KC];    // 2 x 32 KB: Wb tiles
    __shared__ float xsq[TM];
    __shared__ int   idxs[TM];

    const int tid  = threadIdx.x;
    const int lane = tid & 31;
    const int wave = tid >> 5;                      // 0..7, one 16-row M-tile each
    const int l15  = lane & 15;
    const int half = lane >> 4;                     // wave32 lane-half
    const int mblk = blockIdx.x * TM;

    if (tid < TM) xsq[tid] = 0.f;

    // Prefetch B tile for chunk 0 (async: bypasses VGPRs, tracked by ASYNCcnt).
    stage_B(Wb, Blds[0], tid, 0);

    __syncthreads();

    v8f acc[16];
    #pragma unroll
    for (int t = 0; t < 16; ++t) acc[t] = {};

    int pb = 0;
    for (int kc = 0; kc < DPAD; kc += KC) {
        // ---- stage A chunk: X[mblk..+128, kc..+64] f32 -> bf16 LDS, fuse x^2.
        // This VALU/ds work overlaps the in-flight async B copy.
        {
            const int row = tid >> 1;               // 0..127
            const int c0  = (tid & 1) * 32;         // half-row of 32 floats
            unsigned short* dst = &Alds[row * KC + c0];
            float ssum = 0.f;
            if (kc + KC <= DIM) {                   // fast full-width path
                const v4f* src =
                    (const v4f*)(X + (size_t)(mblk + row) * DIM + kc + c0);
                #pragma unroll
                for (int i = 0; i < 8; ++i) {
                    v4f v = __builtin_nontemporal_load(src + i); // stream X
                    ssum += v.x * v.x + v.y * v.y + v.z * v.z + v.w * v.w;
                    dst[i * 4 + 0] = f2bf(v.x);
                    dst[i * 4 + 1] = f2bf(v.y);
                    dst[i * 4 + 2] = f2bf(v.z);
                    dst[i * 4 + 3] = f2bf(v.w);
                }
            } else {                                // tail chunk (cols >= 1000 -> 0)
                #pragma unroll
                for (int i = 0; i < 32; ++i) {
                    const int col = kc + c0 + i;
                    float v = (col < DIM) ? X[(size_t)(mblk + row) * DIM + col] : 0.f;
                    ssum += v * v;
                    dst[i] = f2bf(v);
                }
            }
            atomicAdd(&xsq[row], ssum);             // ds_add_f32
        }

        // ---- prefetch next chunk's B tile into the other buffer, then wait
        // for the current tile only (async ops complete in order: allow the
        // 8 prefetch ops to stay outstanding).
        if (kc + KC < DPAD) {
            stage_B(Wb, Blds[pb ^ 1], tid, kc + KC);
            wait_async<8>();
        } else {
            wait_async<0>();
        }
        __syncthreads();

        // ---- compute: 16 N-tiles x 2 interleaved k-steps of 32 -------------
        const unsigned short* Bcur = Blds[pb];
        const int mrow  = wave * 16;
        const int kb    = half ? 16 : 0;      // A lane-half K base
        const int koff0 = half ? 16 : 0;      // B lane-half K base, ks=0
        const int koff1 = 32 + koff0;         // B lane-half K base, ks=32

        // A fragments for both k-steps (contiguous K order: 16 K per lane-half, matching B).
        FragU a0, a1;
        {
            const unsigned short* ap = &Alds[(mrow + l15) * KC + kb];
            a0.u4[0] = *(const uint4*)(ap);
            a0.u4[1] = *(const uint4*)(ap + 8);
            a1.u4[0] = *(const uint4*)(ap + 32);
            a1.u4[1] = *(const uint4*)(ap + 40);
        }
        // Rotating B fragments, one pair per k-step, so tile t+1's ds_loads
        // can be in flight while tile t's two WMMAs execute.
        FragU b0[2], b1[2];
        {
            const unsigned short* bp = &Bcur[l15 * KC];
            b0[0].u4[0] = *(const uint4*)(bp + koff0);
            b0[0].u4[1] = *(const uint4*)(bp + koff0 + 8);
            b1[0].u4[0] = *(const uint4*)(bp + koff1);
            b1[0].u4[1] = *(const uint4*)(bp + koff1 + 8);
        }
        #pragma unroll
        for (int t = 0; t < 16; ++t) {
            const int cur = t & 1;
            if (t < 15) {
                const unsigned short* bp = &Bcur[((t + 1) * 16 + l15) * KC];
                b0[cur ^ 1].u4[0] = *(const uint4*)(bp + koff0);
                b0[cur ^ 1].u4[1] = *(const uint4*)(bp + koff0 + 8);
                b1[cur ^ 1].u4[0] = *(const uint4*)(bp + koff1);
                b1[cur ^ 1].u4[1] = *(const uint4*)(bp + koff1 + 8);
            }
            acc[t] = __builtin_amdgcn_wmma_f32_16x16x32_bf16(
                false, a0.v, false, b0[cur].v, (short)0, acc[t], false, false);
            acc[t] = __builtin_amdgcn_wmma_f32_16x16x32_bf16(
                false, a1.v, false, b1[cur].v, (short)0, acc[t], false, false);
        }
        // Pipeline spec for this region (between barriers): prime with the A
        // frags + first B pair (8 ds reads), then 2 WMMA overlapped with the
        // next tile's 4 ds reads, repeated. 68 ds reads / 32 WMMA total.
        SGB(0x100, 12, 0);
        #pragma unroll
        for (int g = 0; g < 15; ++g) {
            SGB(0x008, 2, 0);
            SGB(0x100, 4, 0);
        }
        SGB(0x008, 2, 0);

        __syncthreads();
        pb ^= 1;
    }

    // ---- distances + per-row argmin.
    // C/D layout: VGPR r -> M = r (lanes 0-15) or r+8 (lanes 16-31); N = t*16 + (lane&15).
    #pragma unroll
    for (int r = 0; r < 8; ++r) {
        const int m = wave * 16 + r + half * 8;     // local row in [0,128)
        const float xs = xsq[m];
        float best = 3.4e38f;
        int bestn = 0;
        #pragma unroll
        for (int t = 0; t < 16; ++t) {
            const int n = t * 16 + l15;
            const float d = xs - 2.0f * acc[t][r] + wsq[n];
            if (d < best) { best = d; bestn = n; }
        }
        // reduce across the 16 lanes of this half (width-16 keeps halves separate)
        #pragma unroll
        for (int off = 8; off > 0; off >>= 1) {
            const float ob = __shfl_xor(best, off, 16);
            const int   on = __shfl_xor(bestn, off, 16);
            if (ob < best || (ob == best && on < bestn)) { best = ob; bestn = on; }
        }
        if (l15 == 0) idxs[m] = bestn;
    }
    __syncthreads();

    // ---- gather: out[m,:] = W[idxs[m],:]  (250 float4 per row, fp32).
    // W rows are L2-resident; output is streamed (nontemporal).
    for (int r = 0; r < TM; ++r) {
        const int code = idxs[r];
        const v4f* srcp = (const v4f*)(W + (size_t)code * DIM);
        v4f* dstp = (v4f*)(out + (size_t)(mblk + r) * DIM);
        for (int e = tid; e < DIM / 4; e += 256)
            __builtin_nontemporal_store(srcp[e], dstp + e);
    }
}

extern "C" void kernel_launch(void* const* d_in, const int* in_sizes, int n_in,
                              void* d_out, int out_size, void* d_ws, size_t ws_size,
                              hipStream_t stream) {
    const float* X = (const float*)d_in[0];     // (N, D) f32
    const float* W = (const float*)d_in[1];     // (K, D) f32
    float* out = (float*)d_out;                 // (N, D) f32

    unsigned short* Wb = (unsigned short*)d_ws;                       // 512 KB
    float* wsq = (float*)((char*)d_ws + (size_t)KCODES * DPAD * 2);   // 1 KB

    const int N = in_sizes[0] / DIM;            // 65536
    prep_w<<<KCODES, 256, 0, stream>>>(W, Wb, wsq);
    vq_kernel<<<N / TM, 256, 0, stream>>>(X, W, Wb, wsq, out);
}
